// ClusteringLoss_77567109366036
// MI455X (gfx1250) — matrix operand order repaired and checked by reading, not compiled, not run on hardware
//
#include <hip/hip_runtime.h>
#include <hip/hip_bf16.h>
#include <math.h>

// MI455X / gfx1250, wave32. Pipeline:
//  k0: center prep (f16 hi/lo split, |c|^2), inter-cluster loss, zero accums
//  k1: TDM-staged centers -> LDS; WMMA distance GEMM + argmin -> assign[]  (1 feature pass)
//  k2: per-cluster sums/counts via LDS ds_add_f32 + global flush           (1 feature pass)
//  k3: means = sums / max(count,1)
//  k4: deviation norms, per-cluster sum via LDS atomics                    (1 feature pass)
//  k5: final combine -> out[0]

typedef _Float16     v16h __attribute__((ext_vector_type(16)));
typedef float        v8f  __attribute__((ext_vector_type(8)));
typedef unsigned int v4u  __attribute__((ext_vector_type(4)));
typedef int          v4i  __attribute__((ext_vector_type(4)));
typedef int          v8i  __attribute__((ext_vector_type(8)));

#define N_FEAT 262144
#define DIMS   128
#define K_CL   256
#define CH_STRIDE 136   // padded LDS row stride (halfs): 272B rows -> bank spread

union V16HU { v16h v; uint4 q[2]; };

// Tensor Data Mover: 5-arg builtin on ROCm 7.2 (clang-22), 6-arg on clang-23+.
#if defined(__gfx1250__) && __has_builtin(__builtin_amdgcn_tensor_load_to_lds)
#define HAVE_TDM 1
#else
#define HAVE_TDM 0
#endif

// ---------------- ws layout (bytes) ----------------
#define OFF_CHI    0u         // f16 [256][128] center hi
#define OFF_CLO    65536u     // f16 [256][128] center lo
#define OFF_C2     131072u    // f32 [256]      |c|^2
#define OFF_INTER  132096u    // f32 [1]        inter loss
#define OFF_COUNTS 133120u    // f32 [256]
#define OFF_DEV    134144u    // f32 [256]
#define OFF_SUMS   135168u    // f32 [256][128]
#define OFF_MEANS  266240u    // f32 [256][128]
#define OFF_ASSIGN 397312u    // i32 [262144]

// ===================== k0: prep centers + inter loss + zero accums =====================
__global__ void __launch_bounds__(256)
k0_prep(const float* __restrict__ centers,
        _Float16* __restrict__ chi, _Float16* __restrict__ clo,
        float* __restrict__ c2, float* __restrict__ inter,
        float* __restrict__ sums, float* __restrict__ counts, float* __restrict__ devs)
{
    __shared__ float c2s[256];
    __shared__ float red[256];
    const int k = threadIdx.x;

    // hi/lo split + |c|^2
    float ss = 0.f;
    #pragma unroll 4
    for (int d = 0; d < DIMS; ++d) {
        float x = centers[k * DIMS + d];
        ss += x * x;
        _Float16 h = (_Float16)x;
        chi[k * DIMS + d] = h;
        clo[k * DIMS + d] = (_Float16)(x - (float)h);
    }
    c2[k] = ss;
    c2s[k] = ss;
    __syncthreads();

    // strict-upper pairs with dist < MARGIN (dist<1 <=> dist^2<1, incl. clamped negatives)
    float cnt = 0.f;
    for (int i = 0; i < k; ++i) {
        float dot = 0.f;
        #pragma unroll 4
        for (int d = 0; d < DIMS; ++d)
            dot += centers[i * DIMS + d] * centers[k * DIMS + d];
        float d2 = c2s[i] + c2s[k] - 2.f * dot;
        if (d2 < 1.0f) cnt += 1.0f;
    }
    red[k] = cnt;
    __syncthreads();
    for (int s = 128; s > 0; s >>= 1) {
        if (k < s) red[k] += red[k + s];
        __syncthreads();
    }
    if (k == 0) {
        // triu zeros (diag + lower, K*(K+1)/2 = 32896 entries) also count as < margin
        inter[0] = (red[0] + 32896.0f) / 32640.0f;  // denom = K*(K-1)/2
    }

    // zero global accumulators every call
    for (int idx = k; idx < K_CL * DIMS; idx += 256) sums[idx] = 0.f;
    counts[k] = 0.f;
    devs[k]   = 0.f;
}

// ===================== TDM descriptor build (CDNA5 ISA) ============
#if HAVE_TDM
// 2D tile: 256 rows x 64 DWORDs (256B), row stride 64 DWORDs in memory;
// pad_enable inserts 4 DWORDs (16B) after every 64 DWORDs (256B) stored ->
// LDS rows land at the 272B (CH_STRIDE) stride the WMMA B-loads expect.
__device__ __forceinline__ void tdm_load_centers(const void* gptr, unsigned lds_off)
{
    const unsigned long long ga = (unsigned long long)gptr;
    v4u g0;
    g0[0] = 1u;                                              // count=1, is_restore=0, gather=0
    g0[1] = lds_off;                                         // lds_addr (bytes)
    g0[2] = (unsigned)(ga & 0xffffffffu);                    // global_addr[31:0]
    g0[3] = (unsigned)((ga >> 32) & 0x01ffffffu) | (2u << 30);  // addr[56:32] | type=2
    v8i g1;
    g1[0] = (int)((2u << 16)          // data_size = 4B
                | (1u << 20)          // pad_enable
                | (5u << 22)          // pad_interval: 64 DWORDs (256B)
                | (3u << 25));        // pad_amount: 4 DWORDs (16B)
    g1[1] = (int)(64u << 16);         // tensor_dim0 = 64 (bits 79:48)
    g1[2] = (int)(256u << 16);        // tensor_dim1 = 256 (bits 111:80)
    g1[3] = (int)(64u << 16);         // tile_dim0 = 64 (bits 127:112)
    g1[4] = (int)256u;                // tile_dim1 = 256, tile_dim2 = 0
    g1[5] = (int)64u;                 // tensor_dim0_stride = 64 (bits 207:160)
    g1[6] = 0;                        // stride hi / dim1_stride lo
    g1[7] = 0;
    const v4i z4 = {0, 0, 0, 0};      // groups 2/3 unused (2D)
#if __clang_major__ >= 23
    const v8i z8 = {0, 0, 0, 0, 0, 0, 0, 0};
    __builtin_amdgcn_tensor_load_to_lds(g0, g1, z4, z4, z8, 0);   // 6-arg form
#else
    __builtin_amdgcn_tensor_load_to_lds(g0, g1, z4, z4, 0);       // 5-arg form
#endif
}
#endif

// ===================== k1: WMMA distance GEMM + argmin =====================
// grid 2048 x 256 threads (8 waves); wave w handles rows [blk*128 + w*16, +16)
__global__ void __launch_bounds__(256)
k1_assign(const float* __restrict__ feat,
          const _Float16* __restrict__ chi, const _Float16* __restrict__ clo,
          const float* __restrict__ c2, int* __restrict__ assign)
{
    extern __shared__ char smem[];
    _Float16* s_hi = (_Float16*)smem;                    // [256][CH_STRIDE]
    _Float16* s_lo = s_hi + K_CL * CH_STRIDE;            // [256][CH_STRIDE]
    float*    s_c2 = (float*)(s_lo + K_CL * CH_STRIDE);  // [256]

    const int tid = threadIdx.x;

#if HAVE_TDM
    // One wave issues both tensor DMAs (scalar branch via readfirstlane so the
    // other waves genuinely skip issue; TDM ignores EXEC otherwise).
    if (__builtin_amdgcn_readfirstlane((int)tid) == 0) {
        tdm_load_centers(chi, (unsigned)(unsigned long long)s_hi);
        tdm_load_centers(clo, (unsigned)(unsigned long long)s_lo);
        __builtin_amdgcn_s_wait_tensorcnt(0);
    }
    s_c2[tid] = c2[tid];
#else
    {   // fallback: plain uint4 staging with the same padded layout
        const uint4* g_hi = (const uint4*)(chi + tid * DIMS);
        const uint4* g_lo = (const uint4*)(clo + tid * DIMS);
        uint4* d_hi = (uint4*)(s_hi + tid * CH_STRIDE);
        uint4* d_lo = (uint4*)(s_lo + tid * CH_STRIDE);
        #pragma unroll
        for (int j = 0; j < 16; ++j) { d_hi[j] = g_hi[j]; d_lo[j] = g_lo[j]; }
        s_c2[tid] = c2[tid];
    }
#endif
    __syncthreads();

    const int wave = tid >> 5;
    const int lane = tid & 31;
    const int g    = lane >> 4;   // lane group
    const int nl   = lane & 15;
    const int m0   = blockIdx.x * 128 + wave * 16;

    // A fragments (contiguous k order, same as B loads): lane holds row m=nl;
    // elems 0..15 -> K = 32c + 16g + i
    v16h a_hi[4], a_lo[4];
    const float* frow = feat + (size_t)(m0 + nl) * DIMS;
    #pragma unroll
    for (int c = 0; c < 4; ++c) {
        const int base = 32 * c + 16 * g;
        const float4 x0 = *(const float4*)(frow + base);
        const float4 x1 = *(const float4*)(frow + base + 4);
        const float4 y0 = *(const float4*)(frow + base + 8);
        const float4 y1 = *(const float4*)(frow + base + 12);
        const float xv[16] = { x0.x, x0.y, x0.z, x0.w, x1.x, x1.y, x1.z, x1.w,
                               y0.x, y0.y, y0.z, y0.w, y1.x, y1.y, y1.z, y1.w };
        #pragma unroll
        for (int i = 0; i < 16; ++i) {
            _Float16 h = (_Float16)xv[i];
            a_hi[c][i] = h;
            a_lo[c][i] = (_Float16)(xv[i] - (float)h);
        }
    }

    float best[8];
    int   bidx[8];
    #pragma unroll
    for (int r = 0; r < 8; ++r) { best[r] = 3.0e38f; bidx[r] = 0; }

#define LOAD_B(OFFH, H, L) do {                            \
        const uint4* _ph = (const uint4*)(s_hi + (OFFH));  \
        const uint4* _pl = (const uint4*)(s_lo + (OFFH));  \
        (H).q[0] = _ph[0]; (H).q[1] = _ph[1];              \
        (L).q[0] = _pl[0]; (L).q[1] = _pl[1];              \
    } while (0)

    // 16 N-tiles of 16 centers; score = |c|^2 - 2 f.c (|f|^2 constant per row).
    // B loads are software-pipelined one K-chunk ahead of the WMMAs.
    #pragma unroll 1
    for (int t = 0; t < 16; ++t) {
        const int kcol = t * 16 + nl;                 // B: lane holds column n=nl
        const int kb   = kcol * CH_STRIDE + 16 * g;
        V16HU bh0, bl0, bh1, bl1;
        LOAD_B(kb, bh0, bl0);
        v8f acc = {};
        #pragma unroll
        for (int c = 0; c < 4; ++c) {
            if (c < 3) LOAD_B(kb + 32 * (c + 1), bh1, bl1);   // prefetch next chunk
            // near-fp32 dot: hi*hi + lo*hi + hi*lo (drop lo*lo ~ 2^-22 rel)
            acc = __builtin_amdgcn_wmma_f32_16x16x32_f16(false, a_hi[c], false, bh0.v, (short)0, acc, false, false);
            acc = __builtin_amdgcn_wmma_f32_16x16x32_f16(false, a_lo[c], false, bh0.v, (short)0, acc, false, false);
            acc = __builtin_amdgcn_wmma_f32_16x16x32_f16(false, a_hi[c], false, bl0.v, (short)0, acc, false, false);
            if (c < 3) { bh0 = bh1; bl0 = bl1; }
        }
        const float cc = s_c2[kcol];
        #pragma unroll
        for (int r = 0; r < 8; ++r) {                 // D layout: lane holds M = r + 8g
            float score = cc - 2.0f * acc[r];
            if (score < best[r]) { best[r] = score; bidx[r] = kcol; }
        }
    }
#undef LOAD_B

    // min-reduce across the 16 lanes (same g) covering each row's 256 candidates
    #pragma unroll
    for (int r = 0; r < 8; ++r) {
        #pragma unroll
        for (int m = 1; m < 16; m <<= 1) {
            float ov = __shfl_xor(best[r], m, 32);
            int   oi = __shfl_xor(bidx[r], m, 32);
            if (ov < best[r] || (ov == best[r] && oi < bidx[r])) { best[r] = ov; bidx[r] = oi; }
        }
    }
    if (nl == 0) {
        #pragma unroll
        for (int r = 0; r < 8; ++r) assign[m0 + 8 * g + r] = bidx[r];
    }
}

// ===================== k2: per-cluster sums / counts =====================
// grid 256 x 256 threads; LDS accumulators + ds_add_f32, one global-atomic flush per WG
__global__ void __launch_bounds__(256)
k2_accum(const float* __restrict__ feat, const int* __restrict__ assign,
         float* __restrict__ gsums, float* __restrict__ gcounts)
{
    extern __shared__ char smem[];
    float* lsum = (float*)smem;            // [256][132] padded
    float* lcnt = lsum + K_CL * 132;       // [256]

    const int tid = threadIdx.x;
    for (int idx = tid; idx < K_CL * 132; idx += 256) lsum[idx] = 0.f;
    lcnt[tid] = 0.f;
    __syncthreads();

    const int wave = tid >> 5;
    const int lane = tid & 31;
    const int row0 = blockIdx.x * 1024 + wave * 128;
    for (int it = 0; it < 128; ++it) {
        const int row = row0 + it;
        if (it + 8 < 128)   // stream-ahead prefetch (global_prefetch_b8)
            __builtin_prefetch(feat + (size_t)(row + 8) * DIMS + lane * 4, 0, 0);
        const int a = assign[row];
        const float4 v = *(const float4*)(feat + (size_t)row * DIMS + lane * 4);
        float* dst = lsum + a * 132 + lane * 4;
        atomicAdd(dst + 0, v.x);
        atomicAdd(dst + 1, v.y);
        atomicAdd(dst + 2, v.z);
        atomicAdd(dst + 3, v.w);
        if (lane == 0) atomicAdd(&lcnt[a], 1.0f);
    }
    __syncthreads();

    for (int idx = tid; idx < K_CL * DIMS; idx += 256) {
        const float v = lsum[(idx >> 7) * 132 + (idx & 127)];
        if (v != 0.0f) atomicAdd(&gsums[idx], v);
    }
    const float cv = lcnt[tid];
    if (cv != 0.0f) atomicAdd(&gcounts[tid], cv);
}

// ===================== k3: means =====================
__global__ void __launch_bounds__(256)
k3_means(const float* __restrict__ gsums, const float* __restrict__ gcounts,
         float* __restrict__ means)
{
    const int idx = blockIdx.x * 256 + threadIdx.x;   // 32768
    const float sc = fmaxf(gcounts[idx >> 7], 1.0f);
    means[idx] = gsums[idx] / sc;
}

// ===================== k4: deviation norms =====================
__global__ void __launch_bounds__(256)
k4_dev(const float* __restrict__ feat, const int* __restrict__ assign,
       const float* __restrict__ means, float* __restrict__ gdev)
{
    __shared__ float ldev[256];
    const int tid = threadIdx.x;
    ldev[tid] = 0.f;
    __syncthreads();

    const int wave = tid >> 5;
    const int lane = tid & 31;
    const int row0 = blockIdx.x * 1024 + wave * 128;
    for (int it = 0; it < 128; ++it) {
        const int row = row0 + it;
        if (it + 8 < 128)
            __builtin_prefetch(feat + (size_t)(row + 8) * DIMS + lane * 4, 0, 0);
        const int a = assign[row];
        const float4 f = *(const float4*)(feat + (size_t)row * DIMS + lane * 4);
        const float4 m = *(const float4*)(means + a * DIMS + lane * 4);
        const float dx = f.x - m.x, dy = f.y - m.y, dz = f.z - m.z, dw = f.w - m.w;
        float s = dx * dx + dy * dy + dz * dz + dw * dw;
        #pragma unroll
        for (int sh = 16; sh >= 1; sh >>= 1) s += __shfl_xor(s, sh, 32);
        if (lane == 0) atomicAdd(&ldev[a], sqrtf(s));
    }
    __syncthreads();
    const float v = ldev[tid];
    if (v != 0.0f) atomicAdd(&gdev[tid], v);
}

// ===================== k5: final combine =====================
__global__ void __launch_bounds__(256)
k5_final(const float* __restrict__ gcounts, const float* __restrict__ gdev,
         const float* __restrict__ inter, float* __restrict__ out)
{
    __shared__ float rsum[256];
    __shared__ float rval[256];
    const int k = threadIdx.x;
    const float c = gcounts[k];
    const bool valid = c > 1.0f;
    const float pc = gdev[k] / fmaxf(c, 1.0f);
    rsum[k] = valid ? pc : 0.f;
    rval[k] = valid ? 1.f : 0.f;
    __syncthreads();
    for (int s = 128; s > 0; s >>= 1) {
        if (k < s) { rsum[k] += rsum[k + s]; rval[k] += rval[k + s]; }
        __syncthreads();
    }
    if (k == 0) {
        const float nvalid = rval[0];
        const float intra = (nvalid > 0.f) ? (rsum[0] / fmaxf(nvalid, 1.f)) : 0.f;
        out[0] = inter[0] + intra;
    }
}

// ===================== host launcher =====================
extern "C" void kernel_launch(void* const* d_in, const int* in_sizes, int n_in,
                              void* d_out, int out_size, void* d_ws, size_t ws_size,
                              hipStream_t stream)
{
    const float* features = (const float*)d_in[0];   // [262144][128] f32
    const float* centers  = (const float*)d_in[1];   // [256][128]    f32
    float* out = (float*)d_out;

    char* ws = (char*)d_ws;
    _Float16* chi   = (_Float16*)(ws + OFF_CHI);
    _Float16* clo   = (_Float16*)(ws + OFF_CLO);
    float* c2       = (float*)(ws + OFF_C2);
    float* inter    = (float*)(ws + OFF_INTER);
    float* counts   = (float*)(ws + OFF_COUNTS);
    float* devs     = (float*)(ws + OFF_DEV);
    float* sums     = (float*)(ws + OFF_SUMS);
    float* means    = (float*)(ws + OFF_MEANS);
    int*   assign   = (int*)(ws + OFF_ASSIGN);

    const size_t k1_lds = (size_t)2 * K_CL * CH_STRIDE * sizeof(_Float16) + K_CL * sizeof(float); // 140288
    const size_t k2_lds = (size_t)K_CL * 132 * sizeof(float) + K_CL * sizeof(float);              // 136192

    k0_prep  <<<1,    256, 0,      stream>>>(centers, chi, clo, c2, inter, sums, counts, devs);
    k1_assign<<<2048, 256, k1_lds, stream>>>(features, chi, clo, c2, assign);
    k2_accum <<<256,  256, k2_lds, stream>>>(features, assign, sums, counts);
    k3_means <<<128,  256, 0,      stream>>>(sums, counts, means);
    k4_dev   <<<256,  256, 0,      stream>>>(features, assign, means, devs);
    k5_final <<<1,    256, 0,      stream>>>(counts, devs, inter, out);
}
